// Mamba2Simple_29867202576635
// MI455X (gfx1250) — matrix operand order repaired and checked by reading, not compiled, not run on hardware
//
#include <hip/hip_runtime.h>
#include <hip/hip_bf16.h>
#include <math.h>

// ---------------- problem constants ----------------
#define D_MODEL   1024
#define D_STATE   64
#define D_CONV    4
#define HEADDIM   128
#define D_INNER   2048
#define NHEADS    16
#define CONV_DIM  2176      // D_INNER + 2*D_STATE
#define D_IN_PROJ 4240      // 2*D_INNER + 2*D_STATE + NHEADS
#define BATCHSZ   2
#define SEQLEN    2048
#define BL        (BATCHSZ*SEQLEN)   // 4096
#define RMS_EPS   1e-5f

typedef __attribute__((ext_vector_type(16))) _Float16 v16h;
typedef __attribute__((ext_vector_type(8)))  _Float16 v8h;
typedef __attribute__((ext_vector_type(8)))  float    v8f;
typedef __attribute__((ext_vector_type(8)))  float    v8fv;

// ---- gfx1250 async copy global -> LDS (ASYNCcnt path) ----
__device__ __forceinline__ void async_cp_b128(unsigned lds_off, const void* gptr) {
  asm volatile("global_load_async_to_lds_b128 %0, %1, off"
               :: "v"(lds_off), "v"(gptr) : "memory");
}

#if __has_builtin(__builtin_amdgcn_s_wait_asynccnt)
#define WAIT_ASYNC(n) __builtin_amdgcn_s_wait_asynccnt(n)
#else
#define WAIT_ASYNC(n) asm volatile("s_wait_asynccnt %0" :: "n"(n) : "memory")
#endif

__device__ __forceinline__ unsigned lds_off_of(const void* p) {
  return (unsigned)(unsigned long long)
         (__attribute__((address_space(3))) const void*)p;
}

// ---------------- GEMM: C[M,N] = A[M,K] * B[N,K]^T, f16 in / f32 out --------
#define BM 128
#define BN 128
#define BK 32
#define LDS_STR 40   // BK + 8 halves: keeps 16B alignment, dodges bank conflicts
#define NBUF 3
#define TILE_BYTES (BM * LDS_STR * 2)   // 10240 B, same for A and B tiles

__global__ __launch_bounds__(256) void gemm_nt_wmma(
    const _Float16* __restrict__ A,   // [M,K] row-major
    const _Float16* __restrict__ Bw,  // [N,K] row-major
    float* __restrict__ C,            // [M,N] row-major
    int M, int N, int K)
{
  __shared__ __attribute__((aligned(16))) _Float16 sA[NBUF][BM * LDS_STR];
  __shared__ __attribute__((aligned(16))) _Float16 sB[NBUF][BN * LDS_STR];

  const int tid  = threadIdx.x;
  const int lane = tid & 31;
  const int wave = tid >> 5;
  const int wm   = wave & 1;    // 2 wave rows  -> 64 M each
  const int wn   = wave >> 1;   // 4 wave cols  -> 32 N each
  const int bm   = blockIdx.y * BM;
  const int bn   = blockIdx.x * BN;
  const int l15  = lane & 15;
  const int lhi  = lane >> 4;

  // per-thread staging slots: 512 x 16B chunks per tile, 2 per thread.
  // Incrementing global pointers + fixed LDS slot offsets: no per-iter address
  // rebuild, no %NBUF division in the hot loop.
  unsigned slotA[2], slotB[2];
  const _Float16* gA[2];
  const _Float16* gB[2];
  {
    const unsigned sA0 = lds_off_of(&sA[0][0]);
    const unsigned sB0 = lds_off_of(&sB[0][0]);
#pragma unroll
    for (int it = 0; it < 2; ++it) {
      int q   = tid + it * 256;
      int row = q >> 2;
      int off = (q & 3) << 3;
      int rb  = bn + row;
      if (rb >= N) rb = N - 1;   // clamp: junk cols >= N are never stored
      slotA[it] = sA0 + (unsigned)((row * LDS_STR + off) * 2);
      slotB[it] = sB0 + (unsigned)((row * LDS_STR + off) * 2);
      gA[it] = A  + (size_t)(bm + row) * K + off;
      gB[it] = Bw + (size_t)rb * K + off;
    }
  }

  auto issue = [&](unsigned bo) {
#pragma unroll
    for (int it = 0; it < 2; ++it) {
      async_cp_b128(slotA[it] + bo, gA[it]); gA[it] += BK;
      async_cp_b128(slotB[it] + bo, gB[it]); gB[it] += BK;
    }
  };

  v8f acc[4][2];
  {
    v8f z = {};
#pragma unroll
    for (int m = 0; m < 4; ++m)
#pragma unroll
      for (int n = 0; n < 2; ++n) acc[m][n] = z;
  }

  auto compute = [&](unsigned bo) {
    const _Float16* tA = (const _Float16*)((const char*)&sA[0][0] + bo);
    const _Float16* tB = (const _Float16*)((const char*)&sB[0][0] + bo);
    // ---- B fragments: lane l -> col n = l&15, K = (l>>4)*16 + j (contig 32B)
    v16h bfrag[2];
#pragma unroll
    for (int nt = 0; nt < 2; ++nt) {
      const _Float16* src = tB + (wn * 32 + nt * 16 + l15) * LDS_STR + lhi * 16;
#pragma unroll
      for (int j = 0; j < 16; ++j) bfrag[nt][j] = src[j];
    }
    // ---- A fragments + MACs: lane l -> row m = l&15, kbase = (l>>4)*16,
    //      halves 0..15 -> K=kbase+j (contig, matches B)
#pragma unroll
    for (int mt = 0; mt < 4; ++mt) {
      const _Float16* srcA = tA + (wm * 64 + mt * 16 + l15) * LDS_STR + lhi * 16;
      v16h afrag;
#pragma unroll
      for (int j = 0; j < 8; ++j) {
        afrag[j]     = srcA[j];
        afrag[j + 8] = srcA[8 + j];
      }
#pragma unroll
      for (int nt = 0; nt < 2; ++nt) {
        acc[mt][nt] = __builtin_amdgcn_wmma_f32_16x16x32_f16(
            false, afrag, false, bfrag[nt], (short)0, acc[mt][nt], false, false);
      }
    }
  };

  const int nk = K / BK;                 // >= 32 for both GEMMs
  issue(0);
  issue(TILE_BYTES);
  unsigned boIssue = 2 * TILE_BYTES;     // rotating buffer byte-offsets
  unsigned boComp  = 0;

  // steady state: unconditional s_wait_asynccnt 4 (tile i landed, i+1 in flight)
  for (int i = 0; i < nk - 1; ++i) {
    WAIT_ASYNC(4);
    __syncthreads();                     // tile i visible from all waves; all
                                         // readers of the issue buffer are done
    if (i + 2 < nk) {
      issue(boIssue);
      boIssue += TILE_BYTES;
      if (boIssue == NBUF * TILE_BYTES) boIssue = 0;
    }
    compute(boComp);
    boComp += TILE_BYTES;
    if (boComp == NBUF * TILE_BYTES) boComp = 0;
  }
  // epilogue: drain the last tile
  WAIT_ASYNC(0);
  __syncthreads();
  compute(boComp);

  // ---- store: C/D layout: lane l -> col = l&15, VGPR r -> row = (l>>4)*8 + r
#pragma unroll
  for (int mt = 0; mt < 4; ++mt) {
#pragma unroll
    for (int nt = 0; nt < 2; ++nt) {
      int col = bn + wn * 32 + nt * 16 + l15;
      if (col < N) {
#pragma unroll
        for (int r = 0; r < 8; ++r) {
          int row = bm + wm * 64 + mt * 16 + lhi * 8 + r;
          C[(size_t)row * N + col] = acc[mt][nt][r];
        }
      }
    }
  }
}

// ---------------- f32 -> f16 conversion (8 elems / lane) ----------------
__global__ void cvt_f32_f16(const float* __restrict__ in,
                            _Float16* __restrict__ out, int n8) {
  int i = blockIdx.x * blockDim.x + threadIdx.x;
  if (i >= n8) return;
  v8fv a = *(const v8fv*)(in + (size_t)i * 8);
  v8h  h;
#pragma unroll
  for (int j = 0; j < 8; ++j) h[j] = (_Float16)a[j];
  *(v8h*)(out + (size_t)i * 8) = h;
}

// ---------------- dt = softplus(raw + bias) ----------------
__global__ void dt_softplus_kernel(const float* __restrict__ zxbcdt,
                                   const float* __restrict__ dt_bias,
                                   float* __restrict__ dtbuf) {
  int i = blockIdx.x * blockDim.x + threadIdx.x;   // BL*NHEADS
  if (i >= BL * NHEADS) return;
  int h  = i & (NHEADS - 1);
  int bl = i >> 4;
  float x = zxbcdt[(size_t)bl * D_IN_PROJ + (D_INNER + CONV_DIM) + h] + dt_bias[h];
  dtbuf[i] = (x > 20.f) ? x : log1pf(expf(x));
}

// ---------------- causal depthwise conv (w=4) + bias + SiLU ----------------
__global__ void conv_silu_kernel(const float* __restrict__ zxbcdt,
                                 const float* __restrict__ conv_w,
                                 const float* __restrict__ conv_b,
                                 float* __restrict__ xconv) {
  int i = blockIdx.x * blockDim.x + threadIdx.x;   // BL*CONV_DIM
  if (i >= BL * CONV_DIM) return;
  int c  = i % CONV_DIM;
  int bl = i / CONV_DIM;
  int l  = bl % SEQLEN;
  int b  = bl / SEQLEN;
  float acc = conv_b[c];
#pragma unroll
  for (int j = 0; j < D_CONV; ++j) {
    int t = l - (D_CONV - 1) + j;
    if (t >= 0)
      acc = fmaf(conv_w[c * D_CONV + j],
                 zxbcdt[((size_t)b * SEQLEN + t) * D_IN_PROJ + D_INNER + c], acc);
  }
  xconv[i] = acc / (1.f + expf(-acc));             // SiLU
}

// ---------------- selective scan: state[128][64] per (b,h) ----------------
__global__ __launch_bounds__(HEADDIM) void scan_kernel(
    const float* __restrict__ xconv, const float* __restrict__ dtbuf,
    const float* __restrict__ A_log, const float* __restrict__ D_param,
    float* __restrict__ ybuf) {
  const int b = blockIdx.x >> 4;          // NHEADS = 16
  const int h = blockIdx.x & (NHEADS - 1);
  const int p = threadIdx.x;              // 0..127
  const float Ah = -expf(A_log[h]);
  const float Dh = D_param[h];

  float st[D_STATE];
#pragma unroll
  for (int n = 0; n < D_STATE; ++n) st[n] = 0.f;

  __shared__ float Bs[D_STATE];
  __shared__ float Cs[D_STATE];

  for (int t = 0; t < SEQLEN; ++t) {
    size_t row = ((size_t)b * SEQLEN + t) * CONV_DIM;
    if (p < D_STATE)       Bs[p]           = xconv[row + D_INNER + p];
    else                   Cs[p - D_STATE] = xconv[row + D_INNER + p];
    __syncthreads();

    float xt  = xconv[row + h * HEADDIM + p];
    float dtt = dtbuf[((size_t)b * SEQLEN + t) * NHEADS + h];
    float dA  = expf(dtt * Ah);
    float sc  = dtt * xt;
    float y   = 0.f;
#pragma unroll
    for (int n = 0; n < D_STATE; ++n) {
      float s = fmaf(st[n], dA, sc * Bs[n]);
      st[n] = s;
      y = fmaf(s, Cs[n], y);
    }
    ybuf[((size_t)b * SEQLEN + t) * D_INNER + h * HEADDIM + p] = fmaf(Dh, xt, y);
    __syncthreads();
  }
}

// ---------------- gated RMSNorm, emits f16 for out_proj GEMM ----------------
__global__ __launch_bounds__(256) void gated_rmsnorm_kernel(
    const float* __restrict__ ybuf, const float* __restrict__ zxbcdt,
    const float* __restrict__ norm_w, _Float16* __restrict__ yn_h) {
  const int bl = blockIdx.x;
  const float* y = ybuf + (size_t)bl * D_INNER;
  const float* z = zxbcdt + (size_t)bl * D_IN_PROJ;   // z = first D_INNER cols
  __shared__ float red[256];

  float yg[D_INNER / 256];
  float ss = 0.f;
#pragma unroll
  for (int it = 0; it < D_INNER / 256; ++it) {
    int i = threadIdx.x + it * 256;
    float zi = z[i];
    float g  = y[i] * (zi / (1.f + expf(-zi)));
    yg[it] = g;
    ss = fmaf(g, g, ss);
  }
  red[threadIdx.x] = ss;
  __syncthreads();
  for (int s = 128; s > 0; s >>= 1) {
    if (threadIdx.x < s) red[threadIdx.x] += red[threadIdx.x + s];
    __syncthreads();
  }
  float rms = rsqrtf(red[0] * (1.f / D_INNER) + RMS_EPS);
#pragma unroll
  for (int it = 0; it < D_INNER / 256; ++it) {
    int i = threadIdx.x + it * 256;
    yn_h[(size_t)bl * D_INNER + i] = (_Float16)(yg[it] * rms * norm_w[i]);
  }
}

// ---------------- launcher ----------------
extern "C" void kernel_launch(void* const* d_in, const int* in_sizes, int n_in,
                              void* d_out, int out_size, void* d_ws, size_t ws_size,
                              hipStream_t stream) {
  const float* u       = (const float*)d_in[0];
  const float* w1      = (const float*)d_in[1];   // [D_IN_PROJ, D_MODEL]
  const float* conv_w  = (const float*)d_in[2];
  const float* conv_b  = (const float*)d_in[3];
  const float* dt_bias = (const float*)d_in[4];
  const float* A_log   = (const float*)d_in[5];
  const float* D_param = (const float*)d_in[6];
  const float* norm_w  = (const float*)d_in[7];
  const float* w2      = (const float*)d_in[8];   // [D_MODEL, D_INNER]
  float* out = (float*)d_out;

  char* ws = (char*)d_ws;
  size_t off = 0;
  auto take = [&](size_t bytes) -> void* {
    void* p = ws + off;
    off = (off + bytes + 255) & ~(size_t)255;
    return p;
  };
  _Float16* u_h    = (_Float16*)take((size_t)BL * D_MODEL * 2);
  _Float16* w1_h   = (_Float16*)take((size_t)D_IN_PROJ * D_MODEL * 2);
  _Float16* w2_h   = (_Float16*)take((size_t)D_MODEL * D_INNER * 2);
  float*    zxbcdt = (float*)take((size_t)BL * D_IN_PROJ * 4);
  float*    dtbuf  = (float*)take((size_t)BL * NHEADS * 4);
  float*    xconv  = (float*)take((size_t)BL * CONV_DIM * 4);
  float*    ybuf   = (float*)take((size_t)BL * D_INNER * 4);
  _Float16* yn_h   = (_Float16*)take((size_t)BL * D_INNER * 2);
  (void)ws_size; (void)in_sizes; (void)n_in; (void)out_size;

  // f32 -> f16 shadows for WMMA GEMMs
  {
    int n8 = (BL * D_MODEL) / 8;
    cvt_f32_f16<<<(n8 + 255) / 256, 256, 0, stream>>>(u, u_h, n8);
    n8 = (D_IN_PROJ * D_MODEL) / 8;
    cvt_f32_f16<<<(n8 + 255) / 256, 256, 0, stream>>>(w1, w1_h, n8);
    n8 = (D_MODEL * D_INNER) / 8;
    cvt_f32_f16<<<(n8 + 255) / 256, 256, 0, stream>>>(w2, w2_h, n8);
  }

  // in_proj: zxbcdt[BL, 4240] = u[BL,1024] * w1[4240,1024]^T
  {
    dim3 grid((D_IN_PROJ + BN - 1) / BN, BL / BM);
    gemm_nt_wmma<<<grid, 256, 0, stream>>>(u_h, w1_h, zxbcdt, BL, D_IN_PROJ, D_MODEL);
  }

  // dt = softplus(raw + bias)
  {
    int n = BL * NHEADS;
    dt_softplus_kernel<<<(n + 255) / 256, 256, 0, stream>>>(zxbcdt, dt_bias, dtbuf);
  }

  // causal depthwise conv + SiLU over the xBC slice
  {
    int n = BL * CONV_DIM;
    conv_silu_kernel<<<(n + 255) / 256, 256, 0, stream>>>(zxbcdt, conv_w, conv_b, xconv);
  }

  // selective scan: one block per (batch, head)
  scan_kernel<<<BATCHSZ * NHEADS, HEADDIM, 0, stream>>>(xconv, dtbuf, A_log, D_param, ybuf);

  // gated RMSNorm -> f16
  gated_rmsnorm_kernel<<<BL, 256, 0, stream>>>(ybuf, zxbcdt, norm_w, yn_h);

  // out_proj: out[BL,1024] = yn[BL,2048] * w2[1024,2048]^T
  {
    dim3 grid(D_MODEL / BN, BL / BM);
    gemm_nt_wmma<<<grid, 256, 0, stream>>>(yn_h, w2_h, out, BL, D_MODEL, D_INNER);
  }
}
